// ASPwinTransformerBlock3D_21517786153454
// MI455X (gfx1250) — matrix operand order repaired and checked by reading, not compiled, not run on hardware
//
#include <hip/hip_runtime.h>
#include <hip/hip_bf16.h>
#include <cmath>

// ---------------------------------------------------------------------------
// Types / helpers
// ---------------------------------------------------------------------------
typedef __attribute__((ext_vector_type(16))) __bf16 v16bf;
typedef __attribute__((ext_vector_type(8)))  float  v8f;

union FragB {
    unsigned int u[8];
    v16bf        v;
};

__device__ __forceinline__ unsigned short f2b(float f) {
    unsigned int u = __float_as_uint(f);
    unsigned int r = (u + 0x7FFFu + ((u >> 16) & 1u)) >> 16;   // RNE
    return (unsigned short)r;
}
__device__ __forceinline__ float b2f(unsigned short h) {
    return __uint_as_float(((unsigned int)h) << 16);
}
__device__ __forceinline__ float gelu_f(float x) {
    return 0.5f * x * (1.0f + erff(x * 0.70710678118654752f));
}
__device__ __forceinline__ float wsum32(float x) {
    for (int off = 16; off; off >>= 1) x += __shfl_xor(x, off);
    return x;
}
__device__ __forceinline__ v8f wmma_bf16(const FragB& a, const FragB& b, v8f c) {
    return __builtin_amdgcn_wmma_f32_16x16x32_bf16(false, a.v, false, b.v,
                                                   (short)0, c, false, false);
}

// A fragment (16x32 bf16, row-major source, k even, 16B-aligned rows).
// Layout: lane half h=(k/16)&1, vgpr = (k%16)/2  ->  two contiguous uint4 loads.
__device__ __forceinline__ void load_a_row(const unsigned short* A, int ld,
                                           int m0, int k0, int lane, FragB& f) {
    int m = m0 + (lane & 15), h = lane >> 4;
    const uint4* p = (const uint4*)(A + (size_t)m * ld + k0 + 16 * h);
    uint4 x = p[0];
    uint4 y = p[1];                       // +8 elements = +16 bytes
    f.u[0] = x.x; f.u[1] = x.y; f.u[2] = x.z; f.u[3] = x.w;
    f.u[4] = y.x; f.u[5] = y.y; f.u[6] = y.z; f.u[7] = y.w;
}

// B fragment from K-transposed storage: element (k,n) at Bt[n*ld + k].
// Contiguous-k layout (same as A): lane = n + 16*((k/16)&1), vgpr = (k%16)/2  -> 8 contiguous dwords.
__device__ __forceinline__ void load_b_tr(const unsigned short* Bt, int ld,
                                          int n0, int k0, int lane, FragB& f) {
    int n = n0 + (lane & 15), hk = lane >> 4;
    const uint4* p = (const uint4*)(Bt + (size_t)n * ld + k0 + 16 * hk);
    uint4 x = p[0];
    uint4 y = p[1];
    f.u[0] = x.x; f.u[1] = x.y; f.u[2] = x.z; f.u[3] = x.w;
    f.u[4] = y.x; f.u[5] = y.y; f.u[6] = y.z; f.u[7] = y.w;
}

// B fragment from pre-packed global blob (32 lanes x 8 dwords contiguous).
__device__ __forceinline__ void load_b_pk(const unsigned int* Bp, int Nt16,
                                          int kt, int nt, int lane, FragB& f) {
    const uint4* p = (const uint4*)(Bp + (((size_t)kt * Nt16 + nt) * 32 + lane) * 8);
    uint4 x = p[0];
    uint4 y = p[1];
    f.u[0] = x.x; f.u[1] = x.y; f.u[2] = x.z; f.u[3] = x.w;
    f.u[4] = y.x; f.u[5] = y.y; f.u[6] = y.z; f.u[7] = y.w;
}

// ---------------------------------------------------------------------------
// Weight packer: f32 W[K][N] -> WMMA-B bf16 fragments
// blob(kt,nt) = 32 lanes x 8 dwords; dword v of lane l packs
// k = kt*32 + 16*(l>>4) + 2v (+sub), n = nt*16 + (l&15)
// ---------------------------------------------------------------------------
__global__ __launch_bounds__(256) void pack_b_kernel(const float* __restrict__ W,
                                                     int K, int N,
                                                     unsigned int* __restrict__ out) {
    int t = blockIdx.x * 256 + threadIdx.x;
    int total = (K >> 5) * (N >> 4) * 256;
    if (t >= total) return;
    int v = t & 7, lane = (t >> 3) & 31, blob = t >> 8;
    int Nt = N >> 4;
    int nt = blob % Nt, kt = blob / Nt;
    int n = nt * 16 + (lane & 15);
    int k = kt * 32 + 16 * (lane >> 4) + 2 * v;
    unsigned int lo = f2b(W[(size_t)k * N + n]);
    unsigned int hi = f2b(W[(size_t)(k + 1) * N + n]);
    out[t] = lo | (hi << 16);
}

// ---------------------------------------------------------------------------
// LayerNorm over C=96, f32 in -> bf16 out. One wave per row.
// ---------------------------------------------------------------------------
__global__ __launch_bounds__(256) void ln_kernel(const float* __restrict__ x,
                                                 const float* __restrict__ g,
                                                 const float* __restrict__ b,
                                                 unsigned short* __restrict__ out,
                                                 int M) {
    int w = threadIdx.x >> 5, lane = threadIdx.x & 31;
    int row = blockIdx.x * 8 + w;
    if (row >= M) return;
    const float* xr = x + (size_t)row * 96;
    float a0 = xr[lane], a1 = xr[lane + 32], a2 = xr[lane + 64];
    float mean = wsum32(a0 + a1 + a2) * (1.0f / 96.0f);
    float d0 = a0 - mean, d1 = a1 - mean, d2 = a2 - mean;
    float var = wsum32(d0 * d0 + d1 * d1 + d2 * d2) * (1.0f / 96.0f);
    float inv = rsqrtf(var + 1e-5f);
    unsigned short* o = out + (size_t)row * 96;
    o[lane]      = f2b(d0 * inv * g[lane]      + b[lane]);
    o[lane + 32] = f2b(d1 * inv * g[lane + 32] + b[lane + 32]);
    o[lane + 64] = f2b(d2 * inv * g[lane + 64] + b[lane + 64]);
}

// ---------------------------------------------------------------------------
// GEMM: C[M,N] = A[M,K](bf16,row-major) x Bpacked + bias, fused epilogues.
// Block tile 128x96, 8 waves (4x2), wave tile 32x48 (2x3 wmma accumulators).
// ep: 0 out=f32 acc+bias | 1 out=f32 acc+bias+res | 2 out=bf16 gelu(acc+bias)
//     3 out=f32 2*res + gelu(acc+bias)
// ---------------------------------------------------------------------------
__global__ __launch_bounds__(256) void gemm_kernel(const unsigned short* __restrict__ A,
                                                   const unsigned int*  __restrict__ Bp,
                                                   const float* __restrict__ bias,
                                                   const float* __restrict__ resid,
                                                   float* __restrict__ outF,
                                                   unsigned short* __restrict__ outH,
                                                   int M, int N, int K, int ep) {
    int lane = threadIdx.x & 31, w = threadIdx.x >> 5;
    int m0 = blockIdx.x * 128 + (w >> 1) * 32;
    int nBase = blockIdx.y * 96 + (w & 1) * 48;
    int Nt16 = N >> 4;

    v8f acc[2][3];
#pragma unroll
    for (int i = 0; i < 2; ++i)
#pragma unroll
        for (int j = 0; j < 3; ++j)
            acc[i][j] = (v8f){0.f, 0.f, 0.f, 0.f, 0.f, 0.f, 0.f, 0.f};

    int Ksteps = K >> 5;
    for (int kt = 0; kt < Ksteps; ++kt) {
        FragB a[2], b[3];
#pragma unroll
        for (int i = 0; i < 2; ++i)
            load_a_row(A, K, m0 + i * 16, kt * 32, lane, a[i]);
#pragma unroll
        for (int j = 0; j < 3; ++j)
            load_b_pk(Bp, Nt16, kt, (nBase >> 4) + j, lane, b[j]);
#pragma unroll
        for (int i = 0; i < 2; ++i)
#pragma unroll
            for (int j = 0; j < 3; ++j)
                acc[i][j] = wmma_bf16(a[i], b[j], acc[i][j]);
    }

    int h2 = lane >> 4, c0 = lane & 15;
#pragma unroll
    for (int i = 0; i < 2; ++i)
#pragma unroll
        for (int j = 0; j < 3; ++j) {
            int n = nBase + j * 16 + c0;
            float bi = bias[n];
#pragma unroll
            for (int r = 0; r < 8; ++r) {
                int m = m0 + i * 16 + r + 8 * h2;
                size_t o = (size_t)m * N + n;
                float val = acc[i][j][r] + bi;
                if (ep == 0)      outF[o] = val;
                else if (ep == 1) outF[o] = val + resid[o];
                else if (ep == 2) outH[o] = f2b(gelu_f(val));
                else              outF[o] = 2.0f * resid[o] + gelu_f(val);
            }
        }
}

// ---------------------------------------------------------------------------
// Focused linear attention, one block per (window, head).
// kv = k'^T v and out = q' kv via WMMA; 5^3 windowed depthwise conv fused.
// ---------------------------------------------------------------------------
#define NTOK 343
#define NPAD 352

__global__ __launch_bounds__(256) void attn_kernel(const float* __restrict__ qkv,
                                                   const float* __restrict__ pos,
                                                   const float* __restrict__ scale_p,
                                                   const float* __restrict__ dwc_w,
                                                   const float* __restrict__ dwc_b,
                                                   unsigned short* __restrict__ qp,
                                                   unsigned short* __restrict__ attn_a) {
    __shared__ __align__(16) unsigned short s_kT[32 * NPAD];   // k' transposed (c-major)
    __shared__ __align__(16) unsigned short s_vT[32 * NPAD];   // v  transposed
    __shared__ __align__(16) unsigned short s_dw[32 * 128];    // dwc weights bf16
    __shared__ __align__(16) unsigned short s_kvT[32 * 32];    // kv transposed
    __shared__ float s_z[NPAD];
    __shared__ float s_ksum[32];

    int blk = blockIdx.x;                 // 0..1535
    int win = blk / 3, head = blk % 3;
    int nd = win >> 6, nh = (win >> 3) & 7, nw = win & 7;
    int tid = threadIdx.x, lane = tid & 31, w = tid >> 5;
    unsigned short* qpb = qp + (size_t)blk * (NPAD * 32);

    // init: ksum, K-pad columns, q' pad rows, dwc weights
    for (int i = tid; i < 32; i += 256) s_ksum[i] = 0.0f;
    for (int i = tid; i < 32 * 9; i += 256) {
        int c = i / 9, j = NTOK + i % 9;
        s_kT[c * NPAD + j] = 0;
        s_vT[c * NPAD + j] = 0;
    }
    for (int i = tid; i < 9 * 32; i += 256) {
        int j = NTOK + i / 32, c = i % 32;
        qpb[j * 32 + c] = 0;
    }
    for (int i = tid; i < 32 * 125; i += 256) {
        int c = i / 125, t = i % 125;
        s_dw[c * 128 + t] = f2b(dwc_w[c * 125 + t]);
    }
    __syncthreads();

    // ---- phase 0: per-token focusing + transposed LDS stores -------------
    float sc = log1pf(expf(scale_p[lane]));        // softplus
    for (int r = w; r < NTOK; r += 8) {
        int wd = r / 49, wh = (r / 7) % 7, ww = r % 7;
        int vox = ((nd * 7 + wd) * 56 + (nh * 7 + wh)) * 56 + (nw * 7 + ww);
        const float* base = qkv + (size_t)vox * 288 + head * 32 + lane;
        float qv = base[0];
        float kv = base[96] + pos[r * 32 + lane];
        float vv = base[192];
        float qn = (fmaxf(qv, 0.0f) + 1e-6f) / sc;
        float kn = (fmaxf(kv, 0.0f) + 1e-6f) / sc;
        float q3 = qn * qn * qn;
        float k3 = kn * kn * kn;
        float qs1 = wsum32(qn * qn), qs3 = wsum32(q3 * q3);
        float ks1 = wsum32(kn * kn), ks3 = wsum32(k3 * k3);
        float qf = q3 * sqrtf(qs1 / qs3);
        float kf = k3 * sqrtf(ks1 / ks3);
        s_kT[lane * NPAD + r] = f2b(kf);
        s_vT[lane * NPAD + r] = f2b(vv);
        qpb[r * 32 + lane] = f2b(qf);
        atomicAdd(&s_ksum[lane], kf);
    }
    __syncthreads();

    // ---- phase 1: waves 0-3 -> kv = k'^T v (32x32, K=352); waves 4-7 -> z
    if (w < 4) {
        int m0 = (w >> 1) * 16, n0 = (w & 1) * 16;
        v8f acc = (v8f){0.f, 0.f, 0.f, 0.f, 0.f, 0.f, 0.f, 0.f};
        for (int kt = 0; kt < 11; ++kt) {
            FragB a, b;
            load_a_row(s_kT, NPAD, m0, kt * 32, lane, a);
            load_b_tr(s_vT, NPAD, n0, kt * 32, lane, b);
            acc = wmma_bf16(a, b, acc);
        }
        int n = n0 + (lane & 15), h2 = lane >> 4;
#pragma unroll
        for (int r = 0; r < 8; ++r) {
            int m = m0 + r + 8 * h2;
            s_kvT[n * 32 + m] = f2b(acc[r]);       // store transposed for B-frags
        }
    } else {
        for (int r = (w - 4); r < NTOK; r += 4) {
            float qv = b2f(qpb[r * 32 + lane]);
            float d = wsum32(qv * s_ksum[lane]);
            if (lane == 0) s_z[r] = 1.0f / (d + 1e-6f);
        }
    }
    __syncthreads();

    // ---- phase 2: out = q' kv (*z) + dwc(v), write merged-head bf16 ------
    for (int t = w; t < 22; t += 8) {
        int m0 = t * 16;
        FragB a, b0, b1;
        load_a_row(qpb, 32, m0, 0, lane, a);
        load_b_tr(s_kvT, 32, 0, 0, lane, b0);
        load_b_tr(s_kvT, 32, 16, 0, lane, b1);
        v8f zf = (v8f){0.f, 0.f, 0.f, 0.f, 0.f, 0.f, 0.f, 0.f};
        v8f o0 = wmma_bf16(a, b0, zf);
        v8f o1 = wmma_bf16(a, b1, zf);

        int h2 = lane >> 4, c0 = lane & 15;
#pragma unroll
        for (int r = 0; r < 8; ++r) {
            int j = m0 + r + 8 * h2;
            if (j < NTOK) {
                float z = s_z[j];
                int wd = j / 49, wh = (j / 7) % 7, ww = j % 7;
                int vox = ((nd * 7 + wd) * 56 + (nh * 7 + wh)) * 56 + (nw * 7 + ww);
                int kzlo = (wd >= 2) ? -2 : -wd, kzhi = (wd <= 4) ? 2 : 6 - wd;
                int kylo = (wh >= 2) ? -2 : -wh, kyhi = (wh <= 4) ? 2 : 6 - wh;
                int kxlo = (ww >= 2) ? -2 : -ww, kxhi = (ww <= 4) ? 2 : 6 - ww;
#pragma unroll
                for (int f = 0; f < 2; ++f) {
                    int c = c0 + 16 * f;
                    float val = (f ? o1[r] : o0[r]) * z;
                    float conv = dwc_b[c];
                    for (int kz = kzlo; kz <= kzhi; ++kz)
                        for (int ky = kylo; ky <= kyhi; ++ky)
                            for (int kx = kxlo; kx <= kxhi; ++kx) {
                                int jn = j + kz * 49 + ky * 7 + kx;
                                int tap = ((kz + 2) * 5 + (ky + 2)) * 5 + (kx + 2);
                                conv += b2f(s_vT[c * NPAD + jn]) *
                                        b2f(s_dw[c * 128 + tap]);
                            }
                    attn_a[(size_t)vox * 96 + head * 32 + c] = f2b(val + conv);
                }
            }
        }
    }
}

// ---------------------------------------------------------------------------
// MLP 3^3 depthwise conv (full volume, SAME) + exact GELU + per-channel stats
// ---------------------------------------------------------------------------
__global__ __launch_bounds__(256) void conv3_kernel(const unsigned short* __restrict__ xn2,
                                                    const float* __restrict__ cw,
                                                    const float* __restrict__ cb,
                                                    float* __restrict__ y,
                                                    float* __restrict__ sums) {
    __shared__ float s_s[96], s_q[96];
    if (threadIdx.x < 96) { s_s[threadIdx.x] = 0.0f; s_q[threadIdx.x] = 0.0f; }
    __syncthreads();

    size_t idx = (size_t)blockIdx.x * 256 + threadIdx.x;   // exactly 175616*96
    int c = (int)(idx % 96);
    size_t v = idx / 96;
    int Wc = (int)(v % 56), Hc = (int)((v / 56) % 56), Dc = (int)(v / 3136);
    float acc = cb[c];
    int zlo = (Dc > 0) ? -1 : 0, zhi = (Dc < 55) ? 1 : 0;
    int ylo = (Hc > 0) ? -1 : 0, yhi = (Hc < 55) ? 1 : 0;
    int xlo = (Wc > 0) ? -1 : 0, xhi = (Wc < 55) ? 1 : 0;
    for (int kz = zlo; kz <= zhi; ++kz)
        for (int ky = ylo; ky <= yhi; ++ky)
            for (int kx = xlo; kx <= xhi; ++kx) {
                size_t nv = v + (size_t)kz * 3136 + ky * 56 + kx;
                int tap = ((kz + 1) * 3 + (ky + 1)) * 3 + (kx + 1);
                acc += b2f(xn2[nv * 96 + c]) * cw[c * 27 + tap];
            }
    float g = gelu_f(acc);
    y[idx] = g;
    atomicAdd(&s_s[c], g);
    atomicAdd(&s_q[c], g * g);
    __syncthreads();
    if (threadIdx.x < 96) {
        atomicAdd(&sums[threadIdx.x], s_s[threadIdx.x]);
        atomicAdd(&sums[96 + threadIdx.x], s_q[threadIdx.x]);
    }
}

__global__ __launch_bounds__(256) void inorm_kernel(const float* __restrict__ y,
                                                    const float* __restrict__ sums,
                                                    const float* __restrict__ g,
                                                    const float* __restrict__ b,
                                                    unsigned short* __restrict__ out) {
    size_t idx = (size_t)blockIdx.x * 256 + threadIdx.x;
    int c = (int)(idx % 96);
    const float inv_cnt = 1.0f / 175616.0f;
    float m = sums[c] * inv_cnt;
    float var = sums[96 + c] * inv_cnt - m * m;
    out[idx] = f2b((y[idx] - m) * rsqrtf(var + 1e-5f) * g[c] + b[c]);
}

// ---------------------------------------------------------------------------
// Host-side orchestration (graph-capture safe: only launches + memsetAsync)
// ---------------------------------------------------------------------------
extern "C" void kernel_launch(void* const* d_in, const int* in_sizes, int n_in,
                              void* d_out, int out_size, void* d_ws, size_t ws_size,
                              hipStream_t stream) {
    (void)in_sizes; (void)n_in; (void)out_size; (void)ws_size;

    const float* x       = (const float*)d_in[0];
    const float* ln1_g   = (const float*)d_in[1];
    const float* ln1_b   = (const float*)d_in[2];
    const float* qkv_w   = (const float*)d_in[3];
    const float* qkv_b   = (const float*)d_in[4];
    const float* pos_enc = (const float*)d_in[5];
    const float* scale_p = (const float*)d_in[6];
    const float* dwc_w   = (const float*)d_in[7];
    const float* dwc_b   = (const float*)d_in[8];
    const float* proj_w  = (const float*)d_in[9];
    const float* proj_b  = (const float*)d_in[10];
    const float* ln2_g   = (const float*)d_in[11];
    const float* ln2_b   = (const float*)d_in[12];
    const float* conv_w  = (const float*)d_in[13];
    const float* conv_b  = (const float*)d_in[14];
    const float* in_g    = (const float*)d_in[15];
    const float* in_b    = (const float*)d_in[16];
    const float* fc1_w   = (const float*)d_in[17];
    const float* fc1_b   = (const float*)d_in[18];
    const float* fc2_w   = (const float*)d_in[19];
    const float* fc2_b   = (const float*)d_in[20];

    const int M = 175616;                              // 56^3 voxels

    char* ws = (char*)d_ws;
    size_t OFF_QP   = 0;                               // 1536*352*32*2  = 34,603,008
    size_t OFF_XN   = OFF_QP   + 34603008;             // M*96*2         = 33,718,272
    size_t OFF_QKV  = OFF_XN   + 33718272;             // M*288*4        = 202,309,632
    size_t OFF_ATTN = OFF_QKV  + 202309632;            // M*96*2         = 33,718,272
    size_t OFF_X1   = OFF_ATTN + 33718272;             // M*96*4         = 67,436,544
    size_t OFF_SUM  = OFF_X1   + 67436544;             // 768 (pad 1024)
    size_t OFF_PQKV = OFF_SUM  + 1024;                 // 55,296
    size_t OFF_PPRJ = OFF_PQKV + 55296;                // 18,432
    size_t OFF_PFC1 = OFF_PPRJ + 18432;                // 73,728
    size_t OFF_PFC2 = OFF_PFC1 + 73728;                // 73,728

    unsigned short* qp     = (unsigned short*)(ws + OFF_QP);
    unsigned short* xn     = (unsigned short*)(ws + OFF_XN);    // reused for LN2
    float*          qkvbuf = (float*)(ws + OFF_QKV);
    unsigned short* attn   = (unsigned short*)(ws + OFF_ATTN);  // reused as inorm out
    float*          x1     = (float*)(ws + OFF_X1);
    float*          sums   = (float*)(ws + OFF_SUM);
    unsigned int*   pqkv   = (unsigned int*)(ws + OFF_PQKV);
    unsigned int*   pprj   = (unsigned int*)(ws + OFF_PPRJ);
    unsigned int*   pfc1   = (unsigned int*)(ws + OFF_PFC1);
    unsigned int*   pfc2   = (unsigned int*)(ws + OFF_PFC2);
    float*          ybuf   = qkvbuf;                                   // reuse
    unsigned short* hbuf   = (unsigned short*)(ws + OFF_QKV + 67436544); // reuse

    hipMemsetAsync((void*)sums, 0, 768, stream);

    // Pack weights into WMMA B-fragment layout
    pack_b_kernel<<<54, 256, 0, stream>>>(qkv_w, 96, 288, pqkv);
    pack_b_kernel<<<18, 256, 0, stream>>>(proj_w, 96, 96, pprj);
    pack_b_kernel<<<72, 256, 0, stream>>>(fc1_w, 96, 384, pfc1);
    pack_b_kernel<<<72, 256, 0, stream>>>(fc2_w, 384, 96, pfc2);

    // ---- attention branch ----
    ln_kernel<<<21952, 256, 0, stream>>>(x, ln1_g, ln1_b, xn, M);
    gemm_kernel<<<dim3(1372, 3), 256, 0, stream>>>(xn, pqkv, qkv_b, nullptr,
                                                   qkvbuf, nullptr, M, 288, 96, 0);
    attn_kernel<<<1536, 256, 0, stream>>>(qkvbuf, pos_enc, scale_p, dwc_w, dwc_b,
                                          qp, attn);
    gemm_kernel<<<dim3(1372, 1), 256, 0, stream>>>(attn, pprj, proj_b, x,
                                                   x1, nullptr, M, 96, 96, 1);

    // ---- MLP branch ----
    ln_kernel<<<21952, 256, 0, stream>>>(x1, ln2_g, ln2_b, xn, M);
    conv3_kernel<<<65856, 256, 0, stream>>>(xn, conv_w, conv_b, ybuf, sums);
    inorm_kernel<<<65856, 256, 0, stream>>>(ybuf, sums, in_g, in_b, attn);
    gemm_kernel<<<dim3(1372, 4), 256, 0, stream>>>(attn, pfc1, fc1_b, nullptr,
                                                   nullptr, hbuf, M, 384, 96, 2);
    gemm_kernel<<<dim3(1372, 1), 256, 0, stream>>>(hbuf, pfc2, fc2_b, x1,
                                                   (float*)d_out, nullptr,
                                                   M, 96, 384, 3);
}
